// AS_MLP_67886253081286
// MI455X (gfx1250) — matrix operand order repaired and checked by reading, not compiled, not run on hardware
//
#include <hip/hip_runtime.h>

typedef __bf16 bf16_t;
typedef __attribute__((ext_vector_type(8)))  float   v8f;
typedef __attribute__((ext_vector_type(16))) __bf16  v16bf;
typedef __attribute__((ext_vector_type(8)))  __bf16  v8bf;

#define DEV static __device__ __forceinline__

DEV float gelu_f(float x) { return 0.5f * x * (1.0f + erff(x * 0.70710678118654752f)); }

DEV v16bf cat8(v8bf lo, v8bf hi) {
  v16bf r;
#pragma unroll
  for (int i = 0; i < 8; ++i) { r[i] = lo[i]; r[i + 8] = hi[i]; }
  return r;
}

DEV v8f v8f_zero() {
  v8f r;
#pragma unroll
  for (int i = 0; i < 8; ++i) r[i] = 0.0f;
  return r;
}

// Async global -> LDS copy of 16 bytes (per lane), tracked by ASYNCcnt.
// VDST operand = LDS byte offset (flat LDS address truncates to the offset).
DEV void async_copy_b128(const bf16_t* g, bf16_t* l) {
  unsigned lds_off = (unsigned)(size_t)l;
  asm volatile("global_load_async_to_lds_b128 %0, %1, off"
               :: "v"(lds_off), "v"(g) : "memory");
}
DEV void wait_async0() {
  asm volatile("s_wait_asynccnt 0x0" ::: "memory");
}

// ---------------------------------------------------------------------------
// Tiled bf16 WMMA GEMM: out[bt][m][n] = sum_k A[m][k] * Bact[bt][k][n] (+epilogue)
// Block tile: 128(M) x 64(N), K stepped by 32. 256 threads = 8 wave32 waves,
// each wave owns a 32x32 subtile = 2x2 v_wmma_f32_16x16x32_bf16.
// Double-buffered LDS; A tiles staged with global_load_async_to_lds_b128.
// EPI: 0 = +bias -> f32 ; 1 = gelu(acc1+b1)+gelu(acc2+b2) -> f32 (DUAL)
//      2 = +bias+res -> f32 ; 3 = gelu(+bias) -> bf16
// ---------------------------------------------------------------------------
template <int EPI, bool DUAL>
__global__ __launch_bounds__(256)
void gemm_bf16(const bf16_t* __restrict__ A,  const bf16_t* __restrict__ A2,
               const bf16_t* __restrict__ Bm, const bf16_t* __restrict__ Bm2,
               const float* __restrict__ bias, const float* __restrict__ bias2,
               const float* __restrict__ res,
               float* __restrict__ outF, bf16_t* __restrict__ outH,
               int M, int K, int N)
{
  constexpr int NB = DUAL ? 2 : 1;
  __shared__ __align__(16) bf16_t As[2][NB][128 * 32];
  __shared__ __align__(16) bf16_t Bs[2][NB][64 * 32];

  const int bt   = blockIdx.z;
  const int n0   = blockIdx.x * 64;
  const int m0   = blockIdx.y * 128;
  const int tid  = threadIdx.x;
  const int lane = tid & 31;
  const int wave = tid >> 5;
  const int wm   = (wave & 3) * 32;   // 4 waves along M
  const int wn   = (wave >> 2) * 32;  // 2 waves along N

  const size_t bK = (size_t)bt * (size_t)K;
  const int ki = tid >> 3;            // 0..31  (B staging row)
  const int n8 = (tid & 7) * 8;       // 0..56  (B staging col chunk)

  v8f acc[NB][2][2];
#pragma unroll
  for (int s = 0; s < NB; ++s)
#pragma unroll
    for (int mi = 0; mi < 2; ++mi)
#pragma unroll
      for (int ni = 0; ni < 2; ++ni) acc[s][mi][ni] = v8f_zero();

  // ---- staging helpers -----------------------------------------------------
  auto stageA_async = [&](int k0, int buf) {
#pragma unroll
    for (int it = 0; it < 2; ++it) {
      int chunk = tid + it * 256;        // 512 chunks of 8 halves
      int row   = chunk >> 2;            // 0..127
      int c8    = (chunk & 3) * 8;       // 0,8,16,24 halves
      async_copy_b128(A + (size_t)(m0 + row) * K + k0 + c8,
                      &As[buf][0][row * 32 + c8]);
      if constexpr (DUAL)
        async_copy_b128(A2 + (size_t)(m0 + row) * K + k0 + c8,
                        &As[buf][1][row * 32 + c8]);
    }
  };
  auto stageB = [&](int k0, int buf) {   // transpose scatter: Bs[n][k]
    const bf16_t* gb = Bm + (bK + (size_t)(k0 + ki)) * (size_t)N + n0 + n8;
    v8bf vb = *(const v8bf*)gb;
    if constexpr (DUAL) {
      const bf16_t* gb2 = Bm2 + (bK + (size_t)(k0 + ki)) * (size_t)N + n0 + n8;
      v8bf vb2 = *(const v8bf*)gb2;
#pragma unroll
      for (int j = 0; j < 8; ++j) Bs[buf][0][(n8 + j) * 32 + ki] = vb[j];
#pragma unroll
      for (int j = 0; j < 8; ++j) Bs[buf][1][(n8 + j) * 32 + ki] = vb2[j];
    } else {
#pragma unroll
      for (int j = 0; j < 8; ++j) Bs[buf][0][(n8 + j) * 32 + ki] = vb[j];
    }
  };

  // ---- prologue: fill buffer 0 --------------------------------------------
  stageA_async(0, 0);
  stageB(0, 0);
  wait_async0();
  __syncthreads();

  int cur = 0;
  for (int k0 = 0; k0 < K; k0 += 32) {
    const int nxt = cur ^ 1;
    const bool more = (k0 + 32 < K);
    if (more) {
      stageA_async(k0 + 32, nxt);
      stageB(k0 + 32, nxt);
      if (k0 + 64 < K)  // keep a far prefetch in flight -> global_prefetch_b8
        __builtin_prefetch(Bm + (bK + (size_t)(k0 + 64 + ki)) * (size_t)N + n0 + n8, 0, 1);
    }

    // ---- compute from current buffer ----
#pragma unroll
    for (int s = 0; s < NB; ++s) {
      const int ka = (lane < 16) ? 0 : 8;    // A frag: K 0-7&16-23 | 8-15&24-31
      const int kb = (lane < 16) ? 0 : 8;    // B frag: K 0-7&16-23 | 8-15&24-31
      v16bf af[2], bfr[2];
#pragma unroll
      for (int mi = 0; mi < 2; ++mi) {
        int row  = wm + mi * 16 + (lane & 15);
        v8bf lo  = *(const v8bf*)&As[cur][s][row * 32 + ka];
        v8bf hi  = *(const v8bf*)&As[cur][s][row * 32 + ka + 16];
        af[mi]   = cat8(lo, hi);
      }
#pragma unroll
      for (int ni = 0; ni < 2; ++ni) {
        int col  = wn + ni * 16 + (lane & 15);
        v8bf lo  = *(const v8bf*)&Bs[cur][s][col * 32 + kb];
        v8bf hi  = *(const v8bf*)&Bs[cur][s][col * 32 + kb + 16];
        bfr[ni]  = cat8(lo, hi);
      }
#pragma unroll
      for (int mi = 0; mi < 2; ++mi)
#pragma unroll
        for (int ni = 0; ni < 2; ++ni)
          acc[s][mi][ni] = __builtin_amdgcn_wmma_f32_16x16x32_bf16(
              false, af[mi], false, bfr[ni], (short)0, acc[s][mi][ni], false, false);
    }

    if (more) wait_async0();   // next buffer's async A copies landed
    __syncthreads();           // + each wave's B scatters (DScnt) before reuse
    cur = nxt;
  }

  // ---- epilogue ----
  const size_t outBase = (size_t)bt * (size_t)M * (size_t)N;
#pragma unroll
  for (int mi = 0; mi < 2; ++mi) {
#pragma unroll
    for (int ni = 0; ni < 2; ++ni) {
      int n = n0 + wn + ni * 16 + (lane & 15);
#pragma unroll
      for (int j = 0; j < 8; ++j) {
        int m = m0 + wm + mi * 16 + j + ((lane < 16) ? 0 : 8);
        float v = acc[0][mi][ni][j];
        if constexpr (EPI == 1) {
          v = gelu_f(v + bias[m]) + gelu_f(acc[NB - 1][mi][ni][j] + bias2[m]);
        } else {
          v += bias[m];
        }
        size_t idx = outBase + (size_t)m * N + n;
        if constexpr (EPI == 2) v += res[idx];
        if constexpr (EPI == 3) outH[idx] = (bf16_t)gelu_f(v);
        else                    outF[idx] = v;
      }
    }
  }
}

// ---------------------------------------------------------------------------
// GroupNorm(1,C) statistics: deterministic two-stage reduction per batch sample
// ---------------------------------------------------------------------------
__global__ __launch_bounds__(256)
void stats_partial(const float* __restrict__ src, float* __restrict__ part,
                   int perBatch, int S)
{
  int b = blockIdx.y, sb = blockIdx.x;
  int chunk = (perBatch + S - 1) / S;
  int beg = sb * chunk;
  int end = min(beg + chunk, perBatch);
  const float* p = src + (size_t)b * perBatch;
  float s1 = 0.f, s2 = 0.f;
  for (int i = beg + threadIdx.x; i < end; i += 256) {
    float v = p[i];
    s1 += v; s2 += v * v;
  }
  __shared__ float r1[256], r2[256];
  r1[threadIdx.x] = s1; r2[threadIdx.x] = s2;
  __syncthreads();
  for (int st = 128; st > 0; st >>= 1) {
    if (threadIdx.x < st) {
      r1[threadIdx.x] += r1[threadIdx.x + st];
      r2[threadIdx.x] += r2[threadIdx.x + st];
    }
    __syncthreads();
  }
  if (threadIdx.x == 0) {
    part[(b * S + sb) * 2 + 0] = r1[0];
    part[(b * S + sb) * 2 + 1] = r2[0];
  }
}

__global__ __launch_bounds__(64)
void stats_final(const float* __restrict__ part, float* __restrict__ stats,
                 int S, float invN)
{
  int b = blockIdx.x;
  __shared__ float r1[64], r2[64];
  float s1 = 0.f, s2 = 0.f;
  for (int i = threadIdx.x; i < S; i += 64) {
    s1 += part[(b * S + i) * 2 + 0];
    s2 += part[(b * S + i) * 2 + 1];
  }
  r1[threadIdx.x] = s1; r2[threadIdx.x] = s2;
  __syncthreads();
  for (int st = 32; st > 0; st >>= 1) {
    if (threadIdx.x < st) {
      r1[threadIdx.x] += r1[threadIdx.x + st];
      r2[threadIdx.x] += r2[threadIdx.x + st];
    }
    __syncthreads();
  }
  if (threadIdx.x == 0) {
    float m   = r1[0] * invN;
    float var = r2[0] * invN - m * m;
    stats[b * 2 + 0] = m;
    stats[b * 2 + 1] = rsqrtf(var + 1e-5f);
  }
}

// ---------------------------------------------------------------------------
// Apply GN affine (+optional shift-with-zero-pad +GELU) and convert to bf16
// ---------------------------------------------------------------------------
__global__ __launch_bounds__(256)
void prep_norm(const float* __restrict__ src, bf16_t* __restrict__ dst,
               const float* __restrict__ stats, const float* __restrict__ gw,
               const float* __restrict__ gb, int C, int N, int total)
{
  int i = blockIdx.x * 256 + threadIdx.x;
  if (i >= total) return;
  int cn = C * N;
  int b = i / cn;
  int c = (i - b * cn) / N;
  float m = stats[b * 2], r = stats[b * 2 + 1];
  dst[i] = (bf16_t)((src[i] - m) * r * gw[c] + gb[c]);
}

__global__ __launch_bounds__(256)
void prep_shift(const float* __restrict__ src, bf16_t* __restrict__ dst,
                const float* __restrict__ stats, const float* __restrict__ gw,
                const float* __restrict__ gb,
                int C, int H, int W, int dirW, int total)
{
  int i = blockIdx.x * 256 + threadIdx.x;
  if (i >= total) return;
  int w = i % W;
  int h = (i / W) % H;
  int c = (i / (W * H)) % C;
  int b = i / (W * H * C);
  int s = 3 - c / 37;                   // pad - c//ceil(C/7)
  int hh = h, ww = w;
  if (dirW) ww += s; else hh += s;
  float v = 0.f;
  if ((unsigned)hh < (unsigned)H && (unsigned)ww < (unsigned)W) {
    float m = stats[b * 2], r = stats[b * 2 + 1];
    float x = src[((size_t)(b * C + c) * H + hh) * W + ww];
    v = gelu_f((x - m) * r * gw[c] + gb[c]);
  }
  dst[i] = (bf16_t)v;
}

__global__ __launch_bounds__(256)
void wcvt(const float* __restrict__ w, bf16_t* __restrict__ o, int n)
{
  int i = blockIdx.x * 256 + threadIdx.x;
  if (i < n) o[i] = (bf16_t)w[i];
}

// ---------------------------------------------------------------------------
extern "C" void kernel_launch(void* const* d_in, const int* in_sizes, int n_in,
                              void* d_out, int out_size, void* d_ws, size_t ws_size,
                              hipStream_t stream)
{
  (void)in_sizes; (void)n_in; (void)out_size; (void)ws_size;
  const int B = 16, C = 256, H = 56, Wd = 56, HID = 1024;
  const int N = H * Wd;          // 3136
  const int CN = C * N;          // 802816
  const int S = 64;
  const int total = B * CN;      // 12,845,056
  const int totalH = B * HID * N;

  const float* x       = (const float*)d_in[0];
  const float* n1_w    = (const float*)d_in[1];
  const float* n1_b    = (const float*)d_in[2];
  const float* c1_w    = (const float*)d_in[3];
  const float* c1_b    = (const float*)d_in[4];
  const float* an1_w   = (const float*)d_in[5];
  const float* an1_b   = (const float*)d_in[6];
  const float* c21_w   = (const float*)d_in[7];
  const float* c21_b   = (const float*)d_in[8];
  const float* c22_w   = (const float*)d_in[9];
  const float* c22_b   = (const float*)d_in[10];
  const float* an2_w   = (const float*)d_in[11];
  const float* an2_b   = (const float*)d_in[12];
  const float* c3_w    = (const float*)d_in[13];
  const float* c3_b    = (const float*)d_in[14];
  const float* n2_w    = (const float*)d_in[15];
  const float* n2_b    = (const float*)d_in[16];
  const float* fc1_w   = (const float*)d_in[17];
  const float* fc1_b   = (const float*)d_in[18];
  const float* fc2_w   = (const float*)d_in[19];
  const float* fc2_b   = (const float*)d_in[20];

  char* ws = (char*)d_ws;
  size_t off = 0;
  auto take = [&](size_t bytes) -> char* {
    char* p = ws + off;
    off = (off + bytes + 255) & ~(size_t)255;
    return p;
  };
  float*  stats0 = (float*)take(B * 2 * sizeof(float));
  float*  stats1 = (float*)take(B * 2 * sizeof(float));
  float*  stats2 = (float*)take(B * 2 * sizeof(float));
  float*  stats3 = (float*)take(B * 2 * sizeof(float));
  float*  part   = (float*)take((size_t)B * S * 2 * sizeof(float));
  bf16_t* wc1    = (bf16_t*)take((size_t)C * C * 2);
  bf16_t* wc21   = (bf16_t*)take((size_t)C * C * 2);
  bf16_t* wc22   = (bf16_t*)take((size_t)C * C * 2);
  bf16_t* wc3    = (bf16_t*)take((size_t)C * C * 2);
  bf16_t* wf1    = (bf16_t*)take((size_t)HID * C * 2);
  bf16_t* wf2    = (bf16_t*)take((size_t)C * HID * 2);
  bf16_t* bufA   = (bf16_t*)take((size_t)total * 2);   // bf16 activations
  bf16_t* bufT   = (bf16_t*)take((size_t)total * 2);   // bf16 shifted (td)
  float*  bufB   = (float*)take((size_t)total * 4);    // f32 staging (y_raw/z/x2)
  bf16_t* hid    = (bf16_t*)take((size_t)totalH * 2);  // fc1 output

  dim3 blk(256);
  const float invCN = 1.0f / (float)CN;
  const int gTot = (total + 255) / 256;
  dim3 gStat(S, B);
  dim3 gC(N / 64, C / 128, B);     // 49 x 2 x 16
  dim3 gH(N / 64, HID / 128, B);   // 49 x 8 x 16

  // weights -> bf16
  wcvt<<<(C * C + 255) / 256, blk, 0, stream>>>(c1_w, wc1, C * C);
  wcvt<<<(C * C + 255) / 256, blk, 0, stream>>>(c21_w, wc21, C * C);
  wcvt<<<(C * C + 255) / 256, blk, 0, stream>>>(c22_w, wc22, C * C);
  wcvt<<<(C * C + 255) / 256, blk, 0, stream>>>(c3_w, wc3, C * C);
  wcvt<<<(HID * C + 255) / 256, blk, 0, stream>>>(fc1_w, wf1, HID * C);
  wcvt<<<(C * HID + 255) / 256, blk, 0, stream>>>(fc2_w, wf2, C * HID);

  // h = gn(x; n1) -> bf16
  stats_partial<<<gStat, blk, 0, stream>>>(x, part, CN, S);
  stats_final<<<B, 64, 0, stream>>>(part, stats0, S, invCN);
  prep_norm<<<gTot, blk, 0, stream>>>(x, bufA, stats0, n1_w, n1_b, C, N, total);

  // y_raw = c1(h) + b  (f32, for as_n1 stats)
  gemm_bf16<0, false><<<gC, blk, 0, stream>>>(wc1, nullptr, bufA, nullptr,
      c1_b, nullptr, nullptr, bufB, nullptr, C, C, N);

  // t = gelu(gn(y_raw; as_n1)); x_lr = shift_w(t), x_td = shift_h(t) -> bf16
  stats_partial<<<gStat, blk, 0, stream>>>(bufB, part, CN, S);
  stats_final<<<B, 64, 0, stream>>>(part, stats1, S, invCN);
  prep_shift<<<gTot, blk, 0, stream>>>(bufB, bufA, stats1, an1_w, an1_b, C, H, Wd, 1, total);
  prep_shift<<<gTot, blk, 0, stream>>>(bufB, bufT, stats1, an1_w, an1_b, C, H, Wd, 0, total);

  // z = gelu(c21(x_lr)+b21) + gelu(c22(x_td)+b22)  (dual-accumulator GEMM)
  gemm_bf16<1, true><<<gC, blk, 0, stream>>>(wc21, wc22, bufA, bufT,
      c21_b, c22_b, nullptr, bufB, nullptr, C, C, N);

  // zn = gn(z; as_n2) -> bf16 ; x2 = x + c3(zn) + b3
  stats_partial<<<gStat, blk, 0, stream>>>(bufB, part, CN, S);
  stats_final<<<B, 64, 0, stream>>>(part, stats2, S, invCN);
  prep_norm<<<gTot, blk, 0, stream>>>(bufB, bufA, stats2, an2_w, an2_b, C, N, total);
  gemm_bf16<2, false><<<gC, blk, 0, stream>>>(wc3, nullptr, bufA, nullptr,
      c3_b, nullptr, x, bufB, nullptr, C, C, N);

  // MLP: hn = gn(x2; n2) -> bf16 ; hid = gelu(fc1(hn)+b) -> bf16 ; out = x2 + fc2(hid)+b
  stats_partial<<<gStat, blk, 0, stream>>>(bufB, part, CN, S);
  stats_final<<<B, 64, 0, stream>>>(part, stats3, S, invCN);
  prep_norm<<<gTot, blk, 0, stream>>>(bufB, bufA, stats3, n2_w, n2_b, C, N, total);
  gemm_bf16<3, false><<<gH, blk, 0, stream>>>(wf1, nullptr, bufA, nullptr,
      fc1_b, nullptr, nullptr, nullptr, hid, HID, C, N);
  gemm_bf16<2, false><<<gC, blk, 0, stream>>>(wf2, nullptr, hid, nullptr,
      fc2_b, nullptr, bufB, (float*)d_out, nullptr, C, HID, N);
}
